// multiHeadAttention_17025250361592
// MI455X (gfx1250) — matrix operand order repaired and checked by reading, not compiled, not run on hardware
//
#include <hip/hip_runtime.h>
#include <hip/hip_bf16.h>

#define BB 8
#define SS 1024
#define EE 768
#define HH 12
#define DD 64

typedef __attribute__((ext_vector_type(16))) __bf16 v16bf;
typedef __attribute__((ext_vector_type(8)))  float  v8f;

union FragB16 { v16bf v; uint4 q[2]; };

__device__ __forceinline__ __bf16 f2bf(float x) { return (__bf16)x; }

template <typename T> __device__ __forceinline__ T cvtOut(float x);
template <> __device__ __forceinline__ float  cvtOut<float>(float x)  { return x; }
template <> __device__ __forceinline__ __bf16 cvtOut<__bf16>(float x) { return f2bf(x); }

// ---------------- elementwise fp32 -> bf16 ----------------
__global__ void k_cvt(const float* __restrict__ s, __bf16* __restrict__ d, int n) {
  int i = blockIdx.x * 256 + threadIdx.x;
  if (i < n) d[i] = f2bf(s[i]);
}

// src [nh][rows][cols] fp32 -> dst [nh][cols][rows] bf16 (transpose per slab)
__global__ void k_cvt_t(const float* __restrict__ s, __bf16* __restrict__ d,
                        int nh, int rows, int cols) {
  int i = blockIdx.x * 256 + threadIdx.x;
  int n = nh * rows * cols;
  if (i >= n) return;
  int c = i % cols;
  int t = i / cols;
  int r = t % rows;
  int h = t / rows;
  d[((long)h * cols + c) * rows + r] = f2bf(s[i]);
}

// ---------------- generic bf16 GEMM: C = A(MxK,row) * Bt(NxK,row)^T ----------------
// 4 waves/block, wave computes 32x64 (2 M-tiles x 4 N-tiles), K step = 32.
template <typename CT>
__global__ void __launch_bounds__(128)
k_gemm(const __bf16* __restrict__ A, const __bf16* __restrict__ Bt, CT* __restrict__ C,
       int lda, int ldb, int K, long crs, long ccs,
       int batchDiv, long aStride, long bStride, long cStride)
{
  const int lane = threadIdx.x & 31;
  const int wave = threadIdx.x >> 5;
  const int hi   = lane >> 4;   // which 16-lane half
  const int l16  = lane & 15;

  const int bh = blockIdx.y;
  A  += (long)(bh / batchDiv) * aStride;
  Bt += (long)(bh % batchDiv) * bStride;
  C  += (long)bh * cStride;

  const int m0 = blockIdx.x * 128 + wave * 32;
  const int n0 = blockIdx.z * 64;

  v8f acc[2][4];
#pragma unroll
  for (int mt = 0; mt < 2; ++mt)
#pragma unroll
    for (int nt = 0; nt < 4; ++nt)
      acc[mt][nt] = v8f{};

  for (int k0 = 0; k0 < K; k0 += 32) {
    FragB16 a[2];
#pragma unroll
    for (int mt = 0; mt < 2; ++mt) {
      const __bf16* ap = A + (long)(m0 + mt * 16 + l16) * lda + k0 + hi * 8;
      a[mt].q[0] = *(const uint4*)(ap);
      a[mt].q[1] = *(const uint4*)(ap + 16);
    }
#pragma unroll
    for (int nt = 0; nt < 4; ++nt) {
      FragB16 b;
      const __bf16* bp = Bt + (long)(n0 + nt * 16 + l16) * ldb + k0 + hi * 8;
      b.q[0] = *(const uint4*)(bp);
      b.q[1] = *(const uint4*)(bp + 16);
      acc[0][nt] = __builtin_amdgcn_wmma_f32_16x16x32_bf16(
          false, a[0].v, false, b.v, (short)0, acc[0][nt], false, false);
      acc[1][nt] = __builtin_amdgcn_wmma_f32_16x16x32_bf16(
          false, a[1].v, false, b.v, (short)0, acc[1][nt], false, false);
    }
  }

#pragma unroll
  for (int mt = 0; mt < 2; ++mt)
#pragma unroll
    for (int nt = 0; nt < 4; ++nt)
#pragma unroll
      for (int r = 0; r < 8; ++r) {
        long row = m0 + mt * 16 + r + 8 * hi;
        long col = n0 + nt * 16 + l16;
        C[row * crs + col * ccs] = cvtOut<CT>(acc[mt][nt][r]);
      }
}

// ---------------- flash attention per (b,h); wave = 32 query rows ----------------
__global__ void __launch_bounds__(128)
k_flash(const __bf16* __restrict__ Qb, const __bf16* __restrict__ Kb,
        const __bf16* __restrict__ Vt, __bf16* __restrict__ Ob)
{
  const int lane = threadIdx.x & 31;
  const int wave = threadIdx.x >> 5;
  const int hi   = lane >> 4;
  const int l16  = lane & 15;

  const int bh = blockIdx.y;
  const int b  = bh / HH;
  const int h  = bh % HH;

  const __bf16* Q  = Qb + (long)bh * SS * DD;
  const __bf16* Kp = Kb + (long)bh * SS * DD;
  const __bf16* Vp = Vt + (long)bh * DD * SS;   // [d][s]

  __shared__ __align__(16) __bf16 pstage[4 * 1024];  // per-wave 2 x (16x32) bf16 P tiles
  __bf16* pw = pstage + wave * 1024;

  const int s0 = blockIdx.x * 128 + wave * 32;

  // Preload Q fragments: 2 M-tiles x 2 K-steps (d = 0..63)
  FragB16 qf[2][2];
#pragma unroll
  for (int mt = 0; mt < 2; ++mt)
#pragma unroll
    for (int ks = 0; ks < 2; ++ks) {
      const __bf16* qp = Q + (long)(s0 + mt * 16 + l16) * DD + ks * 32 + hi * 8;
      qf[mt][ks].q[0] = *(const uint4*)(qp);
      qf[mt][ks].q[1] = *(const uint4*)(qp + 16);
    }

  v8f oacc[2][4];
  float mrow[2][8], lrow[2][8];
#pragma unroll
  for (int mt = 0; mt < 2; ++mt) {
#pragma unroll
    for (int nt = 0; nt < 4; ++nt) oacc[mt][nt] = v8f{};
#pragma unroll
    for (int r = 0; r < 8; ++r) { mrow[mt][r] = -3.0e38f; lrow[mt][r] = 0.0f; }
  }

  const float scale = 0.03125f * 1.44269504089f;  // (1/sqrt(1024)) * log2(e)

  for (int j = 0; j < SS; j += 32) {
    // S = Q * K^T over this 32-key tile
    v8f sacc[2][2];
    sacc[0][0] = v8f{}; sacc[0][1] = v8f{}; sacc[1][0] = v8f{}; sacc[1][1] = v8f{};
#pragma unroll
    for (int nt = 0; nt < 2; ++nt) {
#pragma unroll
      for (int ks = 0; ks < 2; ++ks) {
        FragB16 kf;
        const __bf16* kp = Kp + (long)(j + nt * 16 + l16) * DD + ks * 32 + hi * 8;
        kf.q[0] = *(const uint4*)(kp);
        kf.q[1] = *(const uint4*)(kp + 16);
        sacc[0][nt] = __builtin_amdgcn_wmma_f32_16x16x32_bf16(
            false, qf[0][ks].v, false, kf.v, (short)0, sacc[0][nt], false, false);
        sacc[1][nt] = __builtin_amdgcn_wmma_f32_16x16x32_bf16(
            false, qf[1][ks].v, false, kf.v, (short)0, sacc[1][nt], false, false);
      }
    }

    // Online softmax update; write P (bf16) into per-wave LDS tiles
#pragma unroll
    for (int mt = 0; mt < 2; ++mt) {
#pragma unroll
      for (int r = 0; r < 8; ++r) {
        float mx = fmaxf(sacc[mt][0][r], sacc[mt][1][r]);
        mx = fmaxf(mx, __shfl_xor(mx, 1));
        mx = fmaxf(mx, __shfl_xor(mx, 2));
        mx = fmaxf(mx, __shfl_xor(mx, 4));
        mx = fmaxf(mx, __shfl_xor(mx, 8));   // reduce across the 16-lane half
        mx *= scale;
        float mnew  = fmaxf(mrow[mt][r], mx);
        float alpha = exp2f(mrow[mt][r] - mnew);
        mrow[mt][r] = mnew;
#pragma unroll
        for (int nt = 0; nt < 4; ++nt) oacc[mt][nt][r] *= alpha;
        float p0 = exp2f(sacc[mt][0][r] * scale - mnew);
        float p1 = exp2f(sacc[mt][1][r] * scale - mnew);
        int row = r + 8 * hi;
        pw[mt * 512 + row * 32 + l16]      = f2bf(p0);
        pw[mt * 512 + row * 32 + 16 + l16] = f2bf(p1);
        float rs = p0 + p1;
        rs += __shfl_xor(rs, 1);
        rs += __shfl_xor(rs, 2);
        rs += __shfl_xor(rs, 4);
        rs += __shfl_xor(rs, 8);
        lrow[mt][r] = lrow[mt][r] * alpha + rs;
      }
    }

    // Re-read P in A-fragment layout (per-wave buffer, wave-order DS ops)
    FragB16 pf[2];
#pragma unroll
    for (int mt = 0; mt < 2; ++mt) {
      const __bf16* pp = pw + mt * 512 + l16 * 32 + hi * 8;
      pf[mt].q[0] = *(const uint4*)(pp);
      pf[mt].q[1] = *(const uint4*)(pp + 16);
    }

    // O += P * V   (V transposed: element (t,d) at Vp[d*SS + t])
#pragma unroll
    for (int dt = 0; dt < 4; ++dt) {
      FragB16 vf;
      const __bf16* vp = Vp + (long)(dt * 16 + l16) * SS + j + hi * 8;
      vf.q[0] = *(const uint4*)(vp);
      vf.q[1] = *(const uint4*)(vp + 16);
      oacc[0][dt] = __builtin_amdgcn_wmma_f32_16x16x32_bf16(
          false, pf[0].v, false, vf.v, (short)0, oacc[0][dt], false, false);
      oacc[1][dt] = __builtin_amdgcn_wmma_f32_16x16x32_bf16(
          false, pf[1].v, false, vf.v, (short)0, oacc[1][dt], false, false);
    }
  }

  // Normalize and store into concat buffer [B,S,H*DD]
#pragma unroll
  for (int mt = 0; mt < 2; ++mt)
#pragma unroll
    for (int r = 0; r < 8; ++r) {
      float inv = 1.0f / lrow[mt][r];
      long row = (long)b * SS + s0 + mt * 16 + r + 8 * hi;
#pragma unroll
      for (int dt = 0; dt < 4; ++dt)
        Ob[row * (HH * DD) + h * DD + dt * 16 + l16] = f2bf(oacc[mt][dt][r] * inv);
    }
}

extern "C" void kernel_launch(void* const* d_in, const int* in_sizes, int n_in,
                              void* d_out, int out_size, void* d_ws, size_t ws_size,
                              hipStream_t stream) {
  (void)in_sizes; (void)n_in; (void)out_size; (void)ws_size;
  const float* x  = (const float*)d_in[0];
  const float* Wk = (const float*)d_in[1];   // note setup_inputs order: x, Wk, Wq, Wv, Wo
  const float* Wq = (const float*)d_in[2];
  const float* Wv = (const float*)d_in[3];
  const float* Wo = (const float*)d_in[4];
  float* out = (float*)d_out;

  char* p = (char*)d_ws;
  __bf16* Xb  = (__bf16*)p; p += (size_t)BB * SS * EE * 2;
  __bf16* Wqt = (__bf16*)p; p += (size_t)HH * DD * EE * 2;
  __bf16* Wkt = (__bf16*)p; p += (size_t)HH * DD * EE * 2;
  __bf16* Wvt = (__bf16*)p; p += (size_t)HH * DD * EE * 2;
  __bf16* Wot = (__bf16*)p; p += (size_t)EE * EE * 2;
  __bf16* Qb  = (__bf16*)p; p += (size_t)BB * HH * SS * DD * 2;
  __bf16* Kb  = (__bf16*)p; p += (size_t)BB * HH * SS * DD * 2;
  __bf16* Vtb = (__bf16*)p; p += (size_t)BB * HH * SS * DD * 2;
  __bf16* Ob  = (__bf16*)p; p += (size_t)BB * SS * HH * DD * 2;

  { int n = BB * SS * EE;
    k_cvt<<<dim3((n + 255) / 256), dim3(256), 0, stream>>>(x, Xb, n); }
  { int n = HH * EE * DD; dim3 g((n + 255) / 256);
    k_cvt_t<<<g, 256, 0, stream>>>(Wq, Wqt, HH, EE, DD);
    k_cvt_t<<<g, 256, 0, stream>>>(Wk, Wkt, HH, EE, DD);
    k_cvt_t<<<g, 256, 0, stream>>>(Wv, Wvt, HH, EE, DD); }
  { int n = EE * EE;
    k_cvt_t<<<dim3((n + 255) / 256), 256, 0, stream>>>(Wo, Wot, 1, EE, EE); }

  // Q,K row-major [S x 64]; V stored transposed [64 x S]
  dim3 gq(SS / 128, BB * HH, 1);
  k_gemm<__bf16><<<gq, 128, 0, stream>>>(Xb, Wqt, Qb, EE, EE, EE,
      (long)DD, 1L, HH, (long)SS * EE, (long)DD * EE, (long)SS * DD);
  k_gemm<__bf16><<<gq, 128, 0, stream>>>(Xb, Wkt, Kb, EE, EE, EE,
      (long)DD, 1L, HH, (long)SS * EE, (long)DD * EE, (long)SS * DD);
  k_gemm<__bf16><<<gq, 128, 0, stream>>>(Xb, Wvt, Vtb, EE, EE, EE,
      1L, (long)SS, HH, (long)SS * EE, (long)DD * EE, (long)SS * DD);

  k_flash<<<dim3(SS / 128, BB * HH), 128, 0, stream>>>(Qb, Kb, Vtb, Ob);

  // out[B*S, E] = Ob[B*S, 768] x Wo[768, E]
  k_gemm<float><<<dim3(BB * SS / 128, 1, EE / 64), 128, 0, stream>>>(
      Ob, Wot, out, HH * DD, EE, HH * DD,
      (long)EE, 1L, 1, 0L, 0L, 0L);
}
